// attn_block_anchor_66357244723394
// MI455X (gfx1250) — matrix operand order repaired and checked by reading, not compiled, not run on hardware
//
#include <hip/hip_runtime.h>

#define TT 4096
#define DD 256

typedef __attribute__((ext_vector_type(16))) __bf16 v16bf;
typedef __attribute__((ext_vector_type(8)))  float  v8f;
typedef __attribute__((ext_vector_type(4)))  unsigned int u32x4;
typedef __attribute__((ext_vector_type(8)))  int i32x8;
typedef __attribute__((ext_vector_type(4)))  int i32x4;

__device__ __forceinline__ float sigmoidf_(float x) { return 1.0f / (1.0f + expf(-x)); }

__device__ __forceinline__ v8f wmma_bf16(v16bf a, v16bf b, v8f c) {
  return __builtin_amdgcn_wmma_f32_16x16x32_bf16(false, a, false, b, (short)0, c, false, false);
}

// A-fragment (16x32 bf16): lane half selects K-half of 16; element s -> K=kh*16+s.
// `row` points at element (m_lane, kbase).
__device__ __forceinline__ v16bf frag_a_f32(const float* row, int kh) {
  v16bf a;
#pragma unroll
  for (int j = 0; j < 4; ++j) {
    int k0 = kh * 16 + 2 * j;
    a[2 * j]     = (__bf16)row[k0];
    a[2 * j + 1] = (__bf16)row[k0 + 1];
    a[8 + 2 * j]     = (__bf16)row[8 + k0];
    a[8 + 2 * j + 1] = (__bf16)row[8 + k0 + 1];
  }
  return a;
}
__device__ __forceinline__ v16bf frag_a_bf16(const __bf16* row, int kh) {
  v16bf a;
#pragma unroll
  for (int j = 0; j < 4; ++j) {
    int k0 = kh * 16 + 2 * j;
    a[2 * j]     = row[k0];
    a[2 * j + 1] = row[k0 + 1];
    a[8 + 2 * j]     = row[8 + k0];
    a[8 + 2 * j + 1] = row[8 + k0 + 1];
  }
  return a;
}
// B-fragment when B^T is row-major (contiguous K): p -> element (n_lane, kbase + 16*kgroup)
__device__ __forceinline__ v16bf frag_bt_f32(const float* p) {
  v16bf b;
#pragma unroll
  for (int e = 0; e < 16; ++e) b[e] = (__bf16)p[e];
  return b;
}
__device__ __forceinline__ v16bf frag_bt_bf16(const __bf16* p) {
  v16bf b;
#pragma unroll
  for (int e = 0; e < 16; ++e) b[e] = p[e];
  return b;
}
// B-fragment when B is row-major KxN (K strided by ld): p -> element (16*kgroup, n_lane)
__device__ __forceinline__ v16bf frag_b_strided(const __bf16* p, int ld) {
  v16bf b;
#pragma unroll
  for (int e = 0; e < 16; ++e) b[e] = p[e * ld];
  return b;
}

// ---------------- Tensor Data Mover: 1-D contiguous tile of `nelem` bf16 -> LDS
// D# per CDNA5 ISA (group0: count/lds/global/type=2,
// group1: data_size=2B, tensor_dim0=1<<30 (no OOB), tile_dim0=nelem, 1-D tile).
// This toolchain exposes the 6-arg builtin: (u32x4, i32x8, i32x4, i32x4, i32x8, i32 cpol).
__device__ __forceinline__ void tdm_load_lds_1d(unsigned lds_addr, const void* gsrc, int nelem) {
  unsigned long long ga = (unsigned long long)gsrc;
  u32x4 g0;
  g0[0] = 1u;                                            // count=1, user mode
  g0[1] = lds_addr;                                      // lds_addr [63:32]
  g0[2] = (unsigned)ga;                                  // global_addr low
  g0[3] = ((unsigned)(ga >> 32) & 0x01FFFFFFu) | (2u << 30);  // addr[56:32] | type=2
  i32x8 g1;
  g1[0] = 1 << 16;          // workgroup_mask=0, data_size=1 (2 bytes)
  g1[1] = 0;                // atomic_barrier_addr=0, tensor_dim0[15:0]=0
  g1[2] = 0x4000 | (1 << 16); // tensor_dim0[31:16]=0x4000 (dim0=1<<30), tensor_dim1 low16=1
  g1[3] = nelem << 16;      // tensor_dim1 high16=0, tile_dim0=nelem
  g1[4] = 0;                // tile_dim1=0 (1-D), tile_dim2=0
  g1[5] = nelem;            // tensor_dim0_stride low32 (benign for 1-D)
  g1[6] = 0;
  g1[7] = 0;
  i32x4 z4; z4[0] = 0; z4[1] = 0; z4[2] = 0; z4[3] = 0;  // groups 2/3 unused (<=2D)
  i32x8 z8;
#pragma unroll
  for (int i = 0; i < 8; ++i) z8[i] = 0;
  __builtin_amdgcn_tensor_load_to_lds(g0, g1, z4, z4, z8, 0);
}

// ---------------- v_gate: sigmoid(Ws@vv+bs) * tanh(Wt@vv+bt), vv = sigmoid(v_vec)
__global__ void gate_kernel(const float* __restrict__ v_vec, const float* __restrict__ Ws,
                            const float* __restrict__ bs, const float* __restrict__ Wt,
                            const float* __restrict__ bt, float* __restrict__ v_gate) {
  __shared__ float vv[DD];
  int e = threadIdx.x;
  vv[e] = sigmoidf_(v_vec[e]);
  __syncthreads();
  float s1 = bs[e], s2 = bt[e];
  for (int d = 0; d < DD; ++d) {
    float g = vv[d];
    s1 += Ws[e * DD + d] * g;
    s2 += Wt[e * DD + d] * g;
  }
  v_gate[e] = sigmoidf_(s1) * tanhf(s2);
}

// ---------------- gated K/V in bf16
__global__ void prep_kernel(const float* __restrict__ value, const float* __restrict__ k_vec,
                            const float* __restrict__ v_gate, __bf16* __restrict__ kw,
                            __bf16* __restrict__ vw, int n) {
  int i = blockIdx.x * blockDim.x + threadIdx.x;
  if (i >= n) return;
  int d = i & (DD - 1);
  float val = value[i];
  kw[i] = (__bf16)(val * sigmoidf_(k_vec[d]));
  vw[i] = (__bf16)(val * v_gate[d]);
}

// ---------------- q = (query @ Wq^T + bq) * sigmoid(q_vec), bf16 out. WMMA GEMM.
__global__ void __launch_bounds__(128)
qproj_kernel(const float* __restrict__ query, const float* __restrict__ Wq,
             const float* __restrict__ bq, const float* __restrict__ q_vec,
             __bf16* __restrict__ qw) {
  const int tid = threadIdx.x;
  const int wave = tid >> 5, lane = tid & 31, ml = lane & 15, kh = lane >> 4;
  const size_t rowb = (size_t)blockIdx.x * 16;  // global row in [0, B*T)

  v8f acc[4];
#pragma unroll
  for (int t = 0; t < 4; ++t)
#pragma unroll
    for (int g = 0; g < 8; ++g) acc[t][g] = 0.0f;

#pragma unroll
  for (int c = 0; c < 8; ++c) {
    v16bf aA = frag_a_f32(query + (rowb + ml) * DD + c * 32, kh);
#pragma unroll
    for (int t = 0; t < 4; ++t) {
      int n0 = wave * 64 + t * 16;
      v16bf bB = frag_bt_f32(Wq + (size_t)(n0 + ml) * DD + c * 32 + kh * 16);
      acc[t] = wmma_bf16(aA, bB, acc[t]);
    }
  }
#pragma unroll
  for (int t = 0; t < 4; ++t) {
    int col = wave * 64 + t * 16 + ml;
    float gate = sigmoidf_(q_vec[col]);
    float bias = bq[col];
#pragma unroll
    for (int g = 0; g < 8; ++g) {
      size_t r = rowb + g + 8 * kh;
      qw[r * DD + col] = (__bf16)((acc[t][g] + bias) * gate);
    }
  }
}

// ---------------- flash attention: 64 rows/block, 4 waves (16 rows each), 32-key blocks.
// K/V tiles double-buffered in LDS, staged by the Tensor Data Mover (wave 0 issues,
// s_wait_tensorcnt + workgroup barrier publishes).
__global__ void __launch_bounds__(128)
flash_kernel(const __bf16* __restrict__ qw, const __bf16* __restrict__ kw,
             const __bf16* __restrict__ vw, float* __restrict__ out) {
  extern __shared__ __align__(16) char smem[];  // 2*16KB K + 2*16KB V + 4KB P = 68KB

  const int tid = threadIdx.x;
  const int wave = tid >> 5, lane = tid & 31, ml = lane & 15, kh = lane >> 4;
  const int b = blockIdx.y;
  const int r0 = blockIdx.x * 64;
  const int rbase = r0 + wave * 16;

  __bf16* Psw = (__bf16*)(smem + 65536) + wave * 512;  // per-wave 16x32 P tile
  const unsigned smem_base = (unsigned)(size_t)smem;   // LDS byte address (addr[31:0])

  // resident Q fragments: 16 rows x 256 cols bf16
  const __bf16* qbase = qw + ((size_t)b * TT + rbase) * DD;
  v16bf aQ[8];
#pragma unroll
  for (int c = 0; c < 8; ++c) aQ[c] = frag_a_bf16(qbase + (size_t)ml * DD + c * 32, kh);

  v8f O[16];
#pragma unroll
  for (int t = 0; t < 16; ++t)
#pragma unroll
    for (int g = 0; g < 8; ++g) O[t][g] = 0.0f;

  float row_m[8], row_l[8];
#pragma unroll
  for (int g = 0; g < 8; ++g) { row_m[g] = -__builtin_inff(); row_l[g] = 0.0f; }

  const int rmax_wave = rbase + 15;
  const int njb = (r0 + 64) / 32;  // causal key blocks for this block
  const float scale = 0.0625f;     // 1/sqrt(256)

  // TDM prefetch of block 0 into buffer 0 (K tile + V tile, 16KB each, contiguous rows)
  if (wave == 0) {
    const size_t gb0 = (size_t)b * TT * DD;
    tdm_load_lds_1d(smem_base, kw + gb0, 32 * DD);
    tdm_load_lds_1d(smem_base + 32768, vw + gb0, 32 * DD);
  }

  for (int j = 0; j < njb; ++j) {
    const int cur = j & 1;
    if (wave == 0) {
      if (j + 1 < njb) {  // prefetch next block into the other buffer, then wait for current
        const size_t gb = ((size_t)b * TT + (size_t)(j + 1) * 32) * DD;
        const unsigned nb = (cur ^ 1) * 16384u;
        tdm_load_lds_1d(smem_base + nb, kw + gb, 32 * DD);
        tdm_load_lds_1d(smem_base + 32768 + nb, vw + gb, 32 * DD);
        __builtin_amdgcn_s_wait_tensorcnt(2);  // in-order: current pair complete
      } else {
        __builtin_amdgcn_s_wait_tensorcnt(0);
      }
    }
    __syncthreads();  // publish current K/V buffer to all waves

    if (j * 32 <= rmax_wave) {
      const __bf16* Ks = (const __bf16*)(smem + cur * 16384);
      const __bf16* Vs = (const __bf16*)(smem + 32768 + cur * 16384);

      // S = Q K^T : two 16x16 f32 tiles over the 32 keys
      v8f s0, s1;
#pragma unroll
      for (int g = 0; g < 8; ++g) { s0[g] = 0.0f; s1[g] = 0.0f; }
#pragma unroll
      for (int c = 0; c < 8; ++c) {
        s0 = wmma_bf16(aQ[c], frag_bt_bf16(Ks + (size_t)ml * DD + c * 32 + kh * 16), s0);
        s1 = wmma_bf16(aQ[c], frag_bt_bf16(Ks + (size_t)(16 + ml) * DD + c * 32 + kh * 16), s1);
      }

      // scale + causal mask with the reference's (w==0 -> -inf) quirk
      const int colb = j * 32;
#pragma unroll
      for (int g = 0; g < 8; ++g) {
        int rg = rbase + g + 8 * kh;
        float x0 = s0[g] * scale, x1 = s1[g] * scale;
        s0[g] = (colb + ml <= rg      && x0 != 0.0f) ? x0 : -__builtin_inff();
        s1[g] = (colb + 16 + ml <= rg && x1 != 0.0f) ? x1 : -__builtin_inff();
      }

      // online softmax (per-row stats replicated across each 16-lane half)
#pragma unroll
      for (int g = 0; g < 8; ++g) {
        float bm = fmaxf(s0[g], s1[g]);
        bm = fmaxf(bm, __shfl_xor(bm, 1, 16));
        bm = fmaxf(bm, __shfl_xor(bm, 2, 16));
        bm = fmaxf(bm, __shfl_xor(bm, 4, 16));
        bm = fmaxf(bm, __shfl_xor(bm, 8, 16));
        float mnew = fmaxf(row_m[g], bm);
        float corr = (mnew == -__builtin_inff()) ? 1.0f : expf(row_m[g] - mnew);
        row_m[g] = mnew;
        float p0 = (s0[g] == -__builtin_inff()) ? 0.0f : expf(s0[g] - mnew);
        float p1 = (s1[g] == -__builtin_inff()) ? 0.0f : expf(s1[g] - mnew);
        float ps = p0 + p1;
        ps += __shfl_xor(ps, 1, 16);
        ps += __shfl_xor(ps, 2, 16);
        ps += __shfl_xor(ps, 4, 16);
        ps += __shfl_xor(ps, 8, 16);
        row_l[g] = row_l[g] * corr + ps;
#pragma unroll
        for (int t = 0; t < 16; ++t) O[t][g] *= corr;
        int prow = g + 8 * kh;  // C/D layout -> row-major P tile in LDS
        Psw[prow * 32 + ml]      = (__bf16)p0;
        Psw[prow * 32 + 16 + ml] = (__bf16)p1;
      }

      // O += P (16x32) @ V (32x16 per n-tile)
      v16bf aP = frag_a_bf16(Psw + (size_t)ml * 32, kh);
#pragma unroll
      for (int t = 0; t < 16; ++t) {
        v16bf bV = frag_b_strided(Vs + (size_t)(kh * 16) * DD + t * 16 + ml, DD);
        O[t] = wmma_bf16(aP, bV, O[t]);
      }
    }
    __syncthreads();  // all waves done with `cur` before it is overwritten
  }

  // epilogue: normalize and store fp32
  float* obase = out + ((size_t)b * TT + rbase) * DD;
#pragma unroll
  for (int g = 0; g < 8; ++g) {
    int r = g + 8 * kh;
    float inv = 1.0f / row_l[g];
#pragma unroll
    for (int t = 0; t < 16; ++t) obase[(size_t)r * DD + t * 16 + ml] = O[t][g] * inv;
  }
}

extern "C" void kernel_launch(void* const* d_in, const int* in_sizes, int n_in,
                              void* d_out, int out_size, void* d_ws, size_t ws_size,
                              hipStream_t stream) {
  const float* query = (const float*)d_in[0];
  const float* value = (const float*)d_in[1];
  const float* q_vec = (const float*)d_in[2];
  const float* k_vec = (const float*)d_in[3];
  const float* v_vec = (const float*)d_in[4];
  const float* Wq = (const float*)d_in[5];
  const float* bq = (const float*)d_in[6];
  const float* Ws = (const float*)d_in[7];
  const float* bs = (const float*)d_in[8];
  const float* Wt = (const float*)d_in[9];
  const float* bt = (const float*)d_in[10];
  float* out = (float*)d_out;

  const size_t NT = (size_t)4 * TT * DD;  // 4,194,304 elems
  char* w = (char*)d_ws;
  float* v_gate = (float*)w;
  __bf16* qw = (__bf16*)(w + 1024);
  __bf16* kw = qw + NT;
  __bf16* vw = kw + NT;

  gate_kernel<<<1, DD, 0, stream>>>(v_vec, Ws, bs, Wt, bt, v_gate);
  prep_kernel<<<(int)((NT + 255) / 256), 256, 0, stream>>>(value, k_vec, v_gate, kw, vw, (int)NT);
  qproj_kernel<<<(4 * TT) / 16, 128, 0, stream>>>(query, Wq, bq, q_vec, qw);
  dim3 grid(TT / 64, 4);
  flash_kernel<<<grid, 128, 69632, stream>>>(qw, kw, vw, out);
}
